// Text2Motion_Transformer_6098853560333
// MI455X (gfx1250) — matrix operand order repaired and checked by reading, not compiled, not run on hardware
//
#include <hip/hip_runtime.h>
#include <hip/hip_bf16.h>
#include <math.h>

// ---------------------------------------------------------------------------
// Text2Motion transformer forward for MI455X (gfx1250, wave32, WMMA).
// GEMMs: 128x128 block tile, 8 waves, each wave = 32x64 output (8 WMMA
// accumulators), bf16 WMMA with f32 accumulate, double-buffered LDS with
// software-pipelined float4 global loads and packed-bf16 ds_store_b64.
// f32->2xbf16 packing is a single v_perm_b32 after RNE rounding adds.
// ---------------------------------------------------------------------------

#define NTOK   6656          // 128 * 52 tokens
#define CDIM   1024
#define FDIM   4096
#define NHEAD  16
#define DHEAD  64
#define SEQ    52
#define BATCH  128

typedef __attribute__((ext_vector_type(16))) __bf16         v16bf;
typedef __attribute__((ext_vector_type(8)))  float          v8f;
typedef __attribute__((ext_vector_type(8)))  unsigned short ushort8;
typedef __attribute__((ext_vector_type(4)))  float          f4;

union BFrag { v16bf v; ushort8 h[2]; };

// Pack two f32 into two bf16 (round-to-nearest-even) in one dword.
// v_perm_b32 selects the high halves of both rounded values in one op:
// sel 0x07060302 -> {src0[31:16], src1[31:16]} with src0=hi, src1=lo.
__device__ __forceinline__ unsigned pack2bf(float lo, float hi) {
  union { float f; unsigned u; } a, b; a.f = lo; b.f = hi;
  const unsigned ua = a.u + 0x7FFFu + ((a.u >> 16) & 1u);
  const unsigned ub = b.u + 0x7FFFu + ((b.u >> 16) & 1u);
  return __builtin_amdgcn_perm(ub, ua, 0x07060302u);
}
__device__ __forceinline__ float gelu_exact(float x) {
  return 0.5f * x * (1.0f + erff(x * 0.70710678118654752f));
}

// ---------------------------------------------------------------------------
// GEMM: out[M,N] = epilogue(A[M,K] @ B[K,N] + bias) (+ residual)
// Requirements honored by call sites: M % 128 == 0, K % 32 == 0.
// NGUARD=true handles ragged N (head, N=1025, also unaligned ldb).
// ---------------------------------------------------------------------------
template<int ACT, bool BIAS, bool RES, bool NGUARD>
__global__ __launch_bounds__(256)
void gemm_wmma_bf16(const float* __restrict__ A, const float* __restrict__ Bm,
                    const float* __restrict__ bias, const float* __restrict__ res,
                    float* __restrict__ out,
                    int M, int N, int K, int lda, int ldb, int ldc) {
  constexpr int LS = 40;     // LDS row stride in halfs: 32 + 8 pad (80 B rows)
  __shared__ __align__(16) unsigned short As[2][128 * LS];  // [m][k] bf16
  __shared__ __align__(16) unsigned short Bs[2][128 * LS];  // [n][k] bf16 (transposed)

  const int tid  = threadIdx.x;
  const int m0   = blockIdx.y * 128;
  const int n0   = blockIdx.x * 128;
  const int lane = tid & 31;
  const int wv   = tid >> 5;        // wave 0..7
  const int wm   = wv >> 1;         // 0..3 : 32-row stripe
  const int wn   = wv & 1;          // 0..1 : 64-col stripe
  const int lr   = lane & 15;
  const int kh   = lane >> 4;       // K-half select per ISA 16-bit layouts

  // --- staging geometry ---
  const int a_row = tid >> 3;         // + it*32 : A rows (128 total)
  const int a_kq  = (tid & 7) * 4;    // A k-quad within 32
  const int b_n   = (tid & 31) * 4;   // B col-quad (128 cols / 4)
  const int b_k   = (tid >> 5) * 4;   // B row-quad (32 k / 4 / 8 waves)

  f4 areg[4];
  f4 breg[4];

  // ---- register loads for K-tile at kk ----
  auto load_regs = [&](int kk) {
    #pragma unroll
    for (int it = 0; it < 4; ++it)
      areg[it] = *(const f4*)&A[(size_t)(m0 + a_row + it * 32) * lda + kk + a_kq];
    if (!NGUARD) {
      #pragma unroll
      for (int j = 0; j < 4; ++j)
        breg[j] = *(const f4*)&Bm[(size_t)(kk + b_k + j) * ldb + n0 + b_n];
    } else {
      #pragma unroll
      for (int j = 0; j < 4; ++j) {
        #pragma unroll
        for (int i = 0; i < 4; ++i) {
          const int n = n0 + b_n + i;
          breg[j][i] = (n < N) ? Bm[(size_t)(kk + b_k + j) * ldb + n] : 0.0f;
        }
      }
    }
  };
  // ---- commit registers to LDS buffer `bf` (packed bf16, ds_store_b64) ----
  auto store_lds = [&](int bf) {
    #pragma unroll
    for (int it = 0; it < 4; ++it) {
      uint2 p;
      p.x = pack2bf(areg[it][0], areg[it][1]);
      p.y = pack2bf(areg[it][2], areg[it][3]);
      *(uint2*)&As[bf][(a_row + it * 32) * LS + a_kq] = p;   // 8B-aligned
    }
    #pragma unroll
    for (int i = 0; i < 4; ++i) {                            // 4x4 transpose
      uint2 p;
      p.x = pack2bf(breg[0][i], breg[1][i]);
      p.y = pack2bf(breg[2][i], breg[3][i]);
      *(uint2*)&Bs[bf][(b_n + i) * LS + b_k] = p;            // 8B-aligned
    }
  };

  v8f acc[2][4] = {};

  load_regs(0);
  store_lds(0);
  __syncthreads();

  int buf = 0;
  for (int kk = 0; kk < K; kk += 32) {
    const bool has_next = (kk + 32 < K);
    if (has_next) load_regs(kk + 32);              // overlap with WMMA burst
    if (kk + 64 < K)                               // speculative far prefetch
      __builtin_prefetch(&A[(size_t)(m0 + a_row) * lda + kk + 64], 0, 0);

    // ---- fragments per ISA 16-bit lane layouts ----
    // A 16x32: lane<16 holds K 0..7 / 16..23, lane>=16 holds K 8..15 / 24..31.
    BFrag fa[2], fb[4];
    #pragma unroll
    for (int sm = 0; sm < 2; ++sm) {
      const unsigned short* ar = &As[buf][(wm * 32 + sm * 16 + lr) * LS + kh * 8];
      fa[sm].h[0] = *(const ushort8*)(ar);
      fa[sm].h[1] = *(const ushort8*)(ar + 16);
    }
    // B 32x16: column N = lane&15; lane<16 K 0..7 / 16..23, lane>=16 K 8..15 / 24..31.
    #pragma unroll
    for (int sn = 0; sn < 4; ++sn) {
      const unsigned short* br = &Bs[buf][(wn * 64 + sn * 16 + lr) * LS + kh * 8];
      fb[sn].h[0] = *(const ushort8*)(br);
      fb[sn].h[1] = *(const ushort8*)(br + 16);
    }
    #pragma unroll
    for (int sm = 0; sm < 2; ++sm)
      #pragma unroll
      for (int sn = 0; sn < 4; ++sn)
        acc[sm][sn] = __builtin_amdgcn_wmma_f32_16x16x32_bf16(
            false, fa[sm].v, false, fb[sn].v, (short)0, acc[sm][sn], false, false);

    if (has_next) store_lds(buf ^ 1);              // fill the other buffer
    __syncthreads();
    buf ^= 1;
  }

  // ---- epilogue: C layout VGPR r -> row = r + kh*8, col = lane&15 ----
  #pragma unroll
  for (int sm = 0; sm < 2; ++sm) {
    #pragma unroll
    for (int sn = 0; sn < 4; ++sn) {
      const int colb = n0 + wn * 64 + sn * 16 + lr;
      #pragma unroll
      for (int r = 0; r < 8; ++r) {
        const int row = m0 + wm * 32 + sm * 16 + kh * 8 + r;
        float v = acc[sm][sn][r];
        if (BIAS)     v += bias[colb];
        if (ACT == 1) v = gelu_exact(v);
        if (RES)      v += res[(size_t)row * ldc + colb];
        if (!NGUARD || colb < N) out[(size_t)row * ldc + colb] = v;
      }
    }
  }
}

// ---------------------------------------------------------------------------
// Token embedding gather + sinusoidal PE (cond row t==0 already holds the
// clip projection written by the preceding GEMM; we just add PE there).
// ---------------------------------------------------------------------------
__global__ __launch_bounds__(256)
void embed_pe_kernel(const int* __restrict__ idx,
                     const float* __restrict__ tok_emb,
                     float* __restrict__ x) {
  const int gid = blockIdx.x * 256 + threadIdx.x;
  if (gid >= NTOK * CDIM) return;
  const int row = gid >> 10;
  const int c   = gid & (CDIM - 1);
  const int b   = row / SEQ;
  const int t   = row % SEQ;
  float v;
  if (t == 0) v = x[gid];
  else        v = tok_emb[(size_t)idx[b * (SEQ - 1) + (t - 1)] * CDIM + c];
  const int   i2  = c >> 1;
  const float dv  = __expf(-(float)(2 * i2) * (9.210340371976184f / (float)CDIM));
  const float ang = (float)t * dv;
  v += (c & 1) ? __cosf(ang) : __sinf(ang);
  x[gid] = v;
}

// ---------------------------------------------------------------------------
// LayerNorm over last dim (1024), one workgroup per row.
// ---------------------------------------------------------------------------
__global__ __launch_bounds__(256)
void ln_kernel(const float* __restrict__ x, const float* __restrict__ w,
               const float* __restrict__ b, float* __restrict__ out) {
  const int row = blockIdx.x;
  const float* xr = x + (size_t)row * CDIM;
  float vals[4];
  float s = 0.0f, s2 = 0.0f;
  #pragma unroll
  for (int i = 0; i < 4; ++i) {
    float v = xr[threadIdx.x + i * 256];
    vals[i] = v; s += v; s2 += v * v;
  }
  __shared__ float r1[256], r2[256];
  r1[threadIdx.x] = s; r2[threadIdx.x] = s2;
  __syncthreads();
  for (int off = 128; off > 0; off >>= 1) {
    if (threadIdx.x < off) {
      r1[threadIdx.x] += r1[threadIdx.x + off];
      r2[threadIdx.x] += r2[threadIdx.x + off];
    }
    __syncthreads();
  }
  const float mean = r1[0] * (1.0f / CDIM);
  const float var  = r2[0] * (1.0f / CDIM) - mean * mean;
  const float rstd = rsqrtf(var + 1e-5f);
  float* orow = out + (size_t)row * CDIM;
  #pragma unroll
  for (int i = 0; i < 4; ++i) {
    const int c = threadIdx.x + i * 256;
    orow[c] = (vals[i] - mean) * rstd * w[c] + b[c];
  }
}

// ---------------------------------------------------------------------------
// Causal attention, one workgroup per (head, batch). T=52, d=64 -> LDS-resident.
// ---------------------------------------------------------------------------
__global__ __launch_bounds__(256)
void attn_kernel(const float* __restrict__ q, const float* __restrict__ k,
                 const float* __restrict__ v, float* __restrict__ y) {
  const int h = blockIdx.x;
  const int b = blockIdx.y;
  __shared__ float qs[SEQ * DHEAD];
  __shared__ float ks[SEQ * DHEAD];
  __shared__ float vs[SEQ * DHEAD];
  __shared__ float sc[SEQ * 56];
  const int tid = threadIdx.x;

  for (int e = tid; e < SEQ * DHEAD; e += 256) {
    const int t = e >> 6, c = e & 63;
    const size_t gi = ((size_t)(b * SEQ + t)) * CDIM + h * DHEAD + c;
    qs[e] = q[gi]; ks[e] = k[gi]; vs[e] = v[gi];
  }
  __syncthreads();

  for (int e = tid; e < SEQ * SEQ; e += 256) {
    const int i = e / SEQ, j = e % SEQ;
    float s;
    if (j <= i) {
      s = 0.0f;
      const float* qi = &qs[i * DHEAD];
      const float* kj = &ks[j * DHEAD];
      #pragma unroll
      for (int c = 0; c < DHEAD; ++c) s += qi[c] * kj[c];
      s *= 0.125f;                       // 1/sqrt(64)
    } else {
      s = -3.0e38f;                      // causal mask
    }
    sc[i * 56 + j] = s;
  }
  __syncthreads();

  if (tid < SEQ) {
    float m = -3.0e38f;
    for (int j = 0; j < SEQ; ++j) m = fmaxf(m, sc[tid * 56 + j]);
    float sum = 0.0f;
    for (int j = 0; j < SEQ; ++j) {
      const float p = __expf(sc[tid * 56 + j] - m);
      sc[tid * 56 + j] = p; sum += p;
    }
    const float inv = 1.0f / sum;
    for (int j = 0; j < SEQ; ++j) sc[tid * 56 + j] *= inv;
  }
  __syncthreads();

  for (int e = tid; e < SEQ * DHEAD; e += 256) {
    const int i = e >> 6, c = e & 63;
    float acc = 0.0f;
    for (int j = 0; j < SEQ; ++j) acc += sc[i * 56 + j] * vs[j * DHEAD + c];
    y[((size_t)(b * SEQ + i)) * CDIM + h * DHEAD + c] = acc;
  }
}

// ---------------------------------------------------------------------------
// Host-side orchestration.
// ---------------------------------------------------------------------------
template<int ACT, bool BIAS, bool RES, bool NGUARD>
static inline void launch_gemm(const float* A, const float* B, const float* bias,
                               const float* res, float* out,
                               int M, int N, int K, int lda, int ldb, int ldc,
                               hipStream_t s) {
  dim3 grid((N + 127) / 128, M / 128);
  gemm_wmma_bf16<ACT, BIAS, RES, NGUARD><<<grid, dim3(256), 0, s>>>(
      A, B, bias, res, out, M, N, K, lda, ldb, ldc);
}

extern "C" void kernel_launch(void* const* d_in, const int* in_sizes, int n_in,
                              void* d_out, int out_size, void* d_ws, size_t ws_size,
                              hipStream_t stream) {
  (void)in_sizes; (void)n_in; (void)out_size; (void)ws_size;

  const int*   idx     = (const int*)  d_in[0];
  const float* clip    = (const float*)d_in[1];
  const float* tok_emb = (const float*)d_in[2];
  const float* cond_w  = (const float*)d_in[3];
  const float* cond_b  = (const float*)d_in[4];
  const float* ln1_w   = (const float*)d_in[5];
  const float* ln1_b   = (const float*)d_in[6];
  const float* wq      = (const float*)d_in[7];
  const float* bq      = (const float*)d_in[8];
  const float* wk      = (const float*)d_in[9];
  const float* bk      = (const float*)d_in[10];
  const float* wvv     = (const float*)d_in[11];
  const float* bv      = (const float*)d_in[12];
  const float* wp      = (const float*)d_in[13];
  const float* bp      = (const float*)d_in[14];
  const float* ln2_w   = (const float*)d_in[15];
  const float* ln2_b   = (const float*)d_in[16];
  const float* w1      = (const float*)d_in[17];
  const float* b1      = (const float*)d_in[18];
  const float* w2      = (const float*)d_in[19];
  const float* b2      = (const float*)d_in[20];
  const float* lnf_w   = (const float*)d_in[21];
  const float* lnf_b   = (const float*)d_in[22];
  const float* head_w  = (const float*)d_in[23];

  // workspace carve-out (floats): x,h,q,k,v,y @ [6656,1024], g @ [6656,4096]
  float* ws = (float*)d_ws;
  const size_t TC = (size_t)NTOK * CDIM;
  float* x  = ws;
  float* hb = x  + TC;
  float* qb = hb + TC;
  float* kb = qb + TC;
  float* vb = kb + TC;
  float* yb = vb + TC;
  float* gb = yb + TC;

  // 1) cond projection -> rows t==0 of x (ldc strides over the 52-token seq)
  launch_gemm<0, true, false, false>(clip, cond_w, cond_b, nullptr, x,
                                     BATCH, CDIM, 512, 512, CDIM, SEQ * CDIM, stream);
  // 2) token embedding gather + sinusoidal PE everywhere
  embed_pe_kernel<<<(NTOK * CDIM + 255) / 256, dim3(256), 0, stream>>>(idx, tok_emb, x);

  for (int i = 0; i < 8; ++i) {
    const size_t oC  = (size_t)i * CDIM;
    const size_t oF  = (size_t)i * FDIM;
    const size_t oCC = (size_t)i * CDIM * CDIM;
    const size_t oCF = (size_t)i * CDIM * FDIM;

    ln_kernel<<<NTOK, dim3(256), 0, stream>>>(x, ln1_w + oC, ln1_b + oC, hb);
    launch_gemm<0, true, false, false>(hb, wq  + oCC, bq + oC, nullptr, qb,
                                       NTOK, CDIM, CDIM, CDIM, CDIM, CDIM, stream);
    launch_gemm<0, true, false, false>(hb, wk  + oCC, bk + oC, nullptr, kb,
                                       NTOK, CDIM, CDIM, CDIM, CDIM, CDIM, stream);
    launch_gemm<0, true, false, false>(hb, wvv + oCC, bv + oC, nullptr, vb,
                                       NTOK, CDIM, CDIM, CDIM, CDIM, CDIM, stream);
    attn_kernel<<<dim3(NHEAD, BATCH), dim3(256), 0, stream>>>(qb, kb, vb, yb);
    launch_gemm<0, true, true,  false>(yb, wp + oCC, bp + oC, x, x,
                                       NTOK, CDIM, CDIM, CDIM, CDIM, CDIM, stream);

    ln_kernel<<<NTOK, dim3(256), 0, stream>>>(x, ln2_w + oC, ln2_b + oC, hb);
    launch_gemm<1, true, false, false>(hb, w1 + oCF, b1 + oF, nullptr, gb,
                                       NTOK, FDIM, CDIM, CDIM, FDIM, FDIM, stream);
    launch_gemm<0, true, true,  false>(gb, w2 + oCF, b2 + oC, x, x,
                                       NTOK, CDIM, FDIM, FDIM, CDIM, CDIM, stream);
  }

  // final LN + head (N = 1025, ragged & unaligned ldb -> guarded path)
  ln_kernel<<<NTOK, dim3(256), 0, stream>>>(x, lnf_w, lnf_b, hb);
  launch_gemm<0, false, false, true>(hb, head_w, nullptr, nullptr, (float*)d_out,
                                     NTOK, 1025, CDIM, CDIM, 1025, 1025, stream);
}
